// CrossScaleAttention_50165218018063
// MI455X (gfx1250) — matrix operand order repaired and checked by reading, not compiled, not run on hardware
//
#include <hip/hip_runtime.h>

#define BATCH 4
#define CH    256      // channels
#define NSP   4096     // H*W = 64*64

typedef __attribute__((ext_vector_type(16))) __bf16 v16bf;
typedef __attribute__((ext_vector_type(8)))  __bf16 v8bf;
typedef __attribute__((ext_vector_type(8)))  float  v8f;
typedef __attribute__((ext_vector_type(4)))  int    v4i;
typedef __attribute__((ext_vector_type(8)))  int    v8i;
typedef __attribute__((ext_vector_type(4)))  unsigned int v4u;

// LDS row strides (elements) padded so 16-lane strided b128 reads are
// bank-conflict-free on the 64x4B banks. TDM pad fields reproduce these:
// K: +4dw per 128dw -> 264-elem rows; V/P: +4dw per 16dw -> 40-elem rows.
#define KSTR 264
#define VSTR 40
#define PSTR 40

#if __has_builtin(__builtin_amdgcn_tensor_load_to_lds) && \
    __has_builtin(__builtin_amdgcn_s_wait_tensorcnt)
#define HAVE_TDM 1
#else
#define HAVE_TDM 0
#endif
#if __has_builtin(__builtin_amdgcn_global_load_async_to_lds_b128)
#define HAVE_ASYNC 1
#else
#define HAVE_ASYNC 0
#endif

static __device__ __forceinline__ v16bf cat8(v8bf lo, v8bf hi) {
  return __builtin_shufflevector(lo, hi, 0,1,2,3,4,5,6,7,8,9,10,11,12,13,14,15);
}

static __device__ __forceinline__ v8f wmma_bf16(v16bf a, v16bf b, v8f c) {
  // D = A(16x32 bf16) * B(32x16 bf16) + C(16x16 f32)
  return __builtin_amdgcn_wmma_f32_16x16x32_bf16(false, a, false, b, (short)0, c,
                                                 false, false);
}

// ---- staging primitives --------------------------------------------------
static __device__ __forceinline__ void cp16_g2l(const __bf16* g, __bf16* l) {
#if HAVE_ASYNC
  __builtin_amdgcn_global_load_async_to_lds_b128(
      (__attribute__((address_space(1))) v4i*)g,
      (__attribute__((address_space(3))) v4i*)l, 0, 0);
#else
  *(v8bf*)l = *(const v8bf*)g;
#endif
}

#if HAVE_TDM
static __device__ __forceinline__ unsigned lds_off(const void* p) {
  return (unsigned)(unsigned long long)(__attribute__((address_space(3))) const void*)p;
}

// D# group1 (ISA 8.4): data_size=2B, pad_enable, 2D tensor (dim2+ unused)
static __device__ __forceinline__ v8i make_g1(unsigned dim0, unsigned dim1,
                                              unsigned tile0, unsigned tile1,
                                              unsigned long long stride0,
                                              unsigned padi, unsigned pada) {
  v8i g;
  g[0] = (int)((1u << 16) | (1u << 20) | (padi << 22) | (pada << 25));
  g[1] = (int)((dim0 & 0xFFFFu) << 16);                       // tensor_dim0 lo
  g[2] = (int)(((dim0 >> 16) & 0xFFFFu) | ((dim1 & 0xFFFFu) << 16));
  g[3] = (int)(((dim1 >> 16) & 0xFFFFu) | ((tile0 & 0xFFFFu) << 16));
  g[4] = (int)(tile1 & 0xFFFFu);                              // tile_dim1; tile_dim2=0
  g[5] = (int)(stride0 & 0xFFFFFFFFu);                        // tensor_dim0_stride
  g[6] = (int)((stride0 >> 32) & 0xFFFFu);                    // dim1_stride = 0
  g[7] = 0;
  return g;
}

// D# group0 (ISA 8.3): count=1, lds_addr, 57b global_addr, type=2 ("image")
static __device__ __forceinline__ void tdm_load_2d(const __bf16* gaddr,
                                                   unsigned lds, v8i g1) {
  unsigned long long ga = (unsigned long long)(size_t)gaddr;
  v4u g0;
  g0[0] = 1u;
  g0[1] = lds;
  g0[2] = (unsigned)ga;
  g0[3] = (unsigned)((ga >> 32) & 0x1FFFFFFu) | (2u << 30);
  v4i gz4 = {0, 0, 0, 0};
  v8i gz8 = {0, 0, 0, 0, 0, 0, 0, 0};
  __builtin_amdgcn_tensor_load_to_lds(g0, g1, gz4, gz4, gz8, 0);
}
#endif

static __device__ __forceinline__ void wait_stage() {
#if HAVE_TDM
  __builtin_amdgcn_s_wait_tensorcnt(0);   // no-op for waves with TENSORcnt==0
#elif HAVE_ASYNC
#if __has_builtin(__builtin_amdgcn_s_wait_asynccnt)
  __builtin_amdgcn_s_wait_asynccnt(0);
#else
  asm volatile("s_wait_asynccnt 0x0" ::: "memory");
#endif
#endif
}

// ---------------------------------------------------------------------------
// Prep: fp32 -> bf16 weight conversion
// ---------------------------------------------------------------------------
__global__ void cvt_f32_bf16(const float* __restrict__ s, __bf16* __restrict__ d, int n) {
  int i = blockIdx.x * blockDim.x + threadIdx.x;
  if (i < n) d[i] = (__bf16)s[i];
}

// Prep: [B,C,N] fp32 -> [B,N,C] bf16 (transpose + convert)
__global__ void transpose_cvt(const float* __restrict__ s, __bf16* __restrict__ d) {
  int i = blockIdx.x * blockDim.x + threadIdx.x;       // over B*C*N
  if (i >= BATCH * CH * NSP) return;
  int b = i / (CH * NSP);
  int r = i - b * CH * NSP;
  int c = r / NSP;
  int n = r - c * NSP;
  d[(size_t)b * NSP * CH + (size_t)n * CH + c] = (__bf16)s[i];
}

// ---------------------------------------------------------------------------
// Projection GEMM: out[o,n] = sum_c W[o,c] * Xt[n,c] + bias[o]
//   STORE_T==1 -> out [B,N,C] (for Q^T / K^T);  STORE_T==0 -> out [B,C,N] (V)
// ---------------------------------------------------------------------------
template <int STORE_T>
__global__ __launch_bounds__(128) void proj_gemm(const __bf16* __restrict__ Xt,
                                                 const __bf16* __restrict__ W,
                                                 const float*  __restrict__ bias,
                                                 __bf16* __restrict__ out) {
  const int lane = threadIdx.x & 31;
  const int l15  = lane & 15;
  const int half = lane >> 4;
  const int wave = blockIdx.x * (blockDim.x >> 5) + (threadIdx.x >> 5);
  const int nt   = wave & (NSP / 16 - 1);
  const int t    = wave >> 8;
  const int ot   = t & (CH / 16 - 1);
  const int b    = t >> 4;

  const __bf16* wr = W + (size_t)(ot * 16 + l15) * CH + half * 16;
  const __bf16* xb = Xt + (size_t)b * NSP * CH + (size_t)(nt * 16 + l15) * CH + half * 16;

  v8f acc = {};
#pragma unroll
  for (int c0 = 0; c0 < CH; c0 += 32) {
    v16bf a  = cat8(*(const v8bf*)(wr + c0), *(const v8bf*)(wr + c0 + 8));
    v16bf bb = *(const v16bf*)(xb + c0);
    acc = wmma_bf16(a, bb, acc);
  }

  const int obase = ot * 16 + half * 8;
#pragma unroll
  for (int m = 0; m < 8; ++m) acc[m] += bias[obase + m];

  if (STORE_T) {
    __bf16* p = out + (size_t)b * NSP * CH + (size_t)(nt * 16 + l15) * CH + obase;
    v8bf st;
#pragma unroll
    for (int m = 0; m < 8; ++m) st[m] = (__bf16)acc[m];
    *(v8bf*)p = st;
  } else {
    __bf16* p = out + (size_t)b * CH * NSP + (size_t)obase * NSP + (nt * 16 + l15);
#pragma unroll
    for (int m = 0; m < 8; ++m) p[(size_t)m * NSP] = (__bf16)acc[m];
  }
}

// ---------------------------------------------------------------------------
// Flash attention, TDM (or async) double-buffered K/V tile staging in LDS.
// Block = 4 waves = 4 consecutive 16-query tiles, shared key-tile schedule.
//   Qt,Kt: [B,N,C] bf16   Vm: [B,C,N] bf16   Ot: [B,N,C] bf16
// ---------------------------------------------------------------------------
__global__ __launch_bounds__(128) void flash_attn(const __bf16* __restrict__ Qt,
                                                  const __bf16* __restrict__ Kt,
                                                  const __bf16* __restrict__ Vm,
                                                  __bf16* __restrict__ Ot) {
  __shared__ __bf16 sK[2][32 * KSTR];   // key-major [j][c], padded rows
  __shared__ __bf16 sV[2][CH * VSTR];   // channel-major [c][j], padded rows
  __shared__ __bf16 sP[4][16 * PSTR];   // per-wave P repack tile

  const int tid  = threadIdx.x;
  const int lane = tid & 31;
  const int l15  = lane & 15;
  const int half = lane >> 4;
  const int w    = tid >> 5;
  const int b    = blockIdx.x >> 6;            // 64 blocks per batch
  const int q0   = ((blockIdx.x & 63) * 4 + w) * 16;

  const __bf16* Qb = Qt + (size_t)b * NSP * CH;
  const __bf16* Kb = Kt + (size_t)b * NSP * CH;
  const __bf16* Vb = Vm + (size_t)b * CH * NSP;
  __bf16* P = sP[w];

#if HAVE_TDM
  // K tile: 32 rows x 256 elems, row stride 256; pad 4dw per 128dw (->KSTR)
  const v8i g1K = make_g1(/*dim0=*/CH, /*dim1=*/32, /*tile0=*/CH, /*tile1=*/32,
                          /*stride0=*/CH, /*padi=*/6, /*pada=*/3);
  // V tile: 256 rows x 32 elems, row stride NSP; pad 4dw per 16dw (->VSTR)
  const v8i g1V = make_g1(/*dim0=*/32, /*dim1=*/CH, /*tile0=*/32, /*tile1=*/CH,
                          /*stride0=*/NSP, /*padi=*/3, /*pada=*/3);
#endif

  auto stage = [&](int j0, int p) {
#if HAVE_TDM
    if (w == 0) {   // one wave drives the Tensor Data Mover for the block
      tdm_load_2d(Kb + (size_t)j0 * CH, lds_off(&sK[p][0]), g1K);
      tdm_load_2d(Vb + j0,              lds_off(&sV[p][0]), g1V);
    }
#else
#pragma unroll
    for (int i = 0; i < 8; ++i) {            // K: 1024 x 16B chunks
      int e = i * 128 + tid;
      int row = e >> 5, col = (e & 31) * 8;
      cp16_g2l(Kb + (size_t)(j0 + row) * CH + col, &sK[p][row * KSTR + col]);
    }
#pragma unroll
    for (int i = 0; i < 8; ++i) {            // V: 1024 x 16B chunks
      int e = i * 128 + tid;
      int c = e >> 2, k = (e & 3) * 8;
      cp16_g2l(Vb + (size_t)c * NSP + j0 + k, &sV[p][c * VSTR + k]);
    }
#endif
  };

  // Preload Q A-fragments (16 queries x full C=256), reused for every j step
  v16bf qa[8];
#pragma unroll
  for (int kt = 0; kt < 8; ++kt) {
    const __bf16* qp = Qb + (size_t)(q0 + l15) * CH + kt * 32 + half * 16;
    qa[kt] = cat8(*(const v8bf*)qp, *(const v8bf*)(qp + 8));
  }

  v8f oacc[16];
  v8f zero = {};
#pragma unroll
  for (int ct = 0; ct < 16; ++ct) oacc[ct] = zero;
  float mrow[8], lrow[8];
#pragma unroll
  for (int m = 0; m < 8; ++m) { mrow[m] = -3.4e38f; lrow[m] = 0.0f; }

  stage(0, 0);

  for (int it = 0; it < NSP / 32; ++it) {
    const int p  = it & 1;
    const int j0 = it * 32;

    wait_stage();         // buffer p resident (issuing wave waits the counter)
    __syncthreads();      // all waves see it; prev buffer is free to overwrite
    if (it + 1 < NSP / 32) stage(j0 + 32, p ^ 1);   // overlap next fetch

    // ---- S tiles: 16 queries x 32 keys (software-pipelined LDS reads) ----
    const __bf16* kc = sK[p] + l15 * KSTR + half * 16;
    v8f s0 = zero, s1 = zero;
    v16bf kb0 = *(const v16bf*)(kc);
    v16bf kb1 = *(const v16bf*)(kc + 16 * KSTR);
#pragma unroll
    for (int kt = 0; kt < 8; ++kt) {
      v16bf kn0, kn1;
      if (kt < 7) {
        kn0 = *(const v16bf*)(kc + (kt + 1) * 32);
        kn1 = *(const v16bf*)(kc + (kt + 1) * 32 + 16 * KSTR);
      }
      s0 = wmma_bf16(qa[kt], kb0, s0);
      s1 = wmma_bf16(qa[kt], kb1, s1);
      kb0 = kn0; kb1 = kn1;
    }

    // ---- online softmax (row r = m + half*8; cols across 16 lanes) ----
    float scale[8];
#pragma unroll
    for (int m = 0; m < 8; ++m) {
      float rm = fmaxf(s0[m], s1[m]);
      rm = fmaxf(rm, __shfl_xor(rm, 1, 32));
      rm = fmaxf(rm, __shfl_xor(rm, 2, 32));
      rm = fmaxf(rm, __shfl_xor(rm, 4, 32));
      rm = fmaxf(rm, __shfl_xor(rm, 8, 32));
      float mn = fmaxf(mrow[m], rm);
      scale[m] = __expf(mrow[m] - mn);
      s0[m] = __expf(s0[m] - mn);
      s1[m] = __expf(s1[m] - mn);
      float rs = s0[m] + s1[m];
      rs += __shfl_xor(rs, 1, 32);
      rs += __shfl_xor(rs, 2, 32);
      rs += __shfl_xor(rs, 4, 32);
      rs += __shfl_xor(rs, 8, 32);
      lrow[m] = lrow[m] * scale[m] + rs;
      mrow[m] = mn;
    }
#pragma unroll
    for (int ct = 0; ct < 16; ++ct)
#pragma unroll
      for (int m = 0; m < 8; ++m) oacc[ct][m] *= scale[m];

    // ---- repack P (C/D layout) -> A-fragment layout through LDS ----
#pragma unroll
    for (int m = 0; m < 8; ++m) {
      P[(m + half * 8) * PSTR + l15]      = (__bf16)s0[m];
      P[(m + half * 8) * PSTR + 16 + l15] = (__bf16)s1[m];
    }
    __builtin_amdgcn_wave_barrier();
    asm volatile("s_wait_dscnt 0x0" ::: "memory");  // cross-lane LDS RAW in-wave
    const __bf16* pp = P + l15 * PSTR + half * 16;
    v16bf pa = cat8(*(const v8bf*)pp, *(const v8bf*)(pp + 8));

    // ---- O += P * V^T : 16 channel tiles, K = 32 keys (pipelined) ----
    const __bf16* vc = sV[p] + l15 * VSTR + half * 16;
    v16bf vb = *(const v16bf*)(vc);
#pragma unroll
    for (int ct = 0; ct < 16; ++ct) {
      v16bf vn;
      if (ct < 15) vn = *(const v16bf*)(vc + (ct + 1) * 16 * VSTR);
      oacc[ct] = wmma_bf16(pa, vb, oacc[ct]);
      vb = vn;
    }
  }

  // ---- normalize and store Ot [B,N,C] ----
  float inv[8];
#pragma unroll
  for (int m = 0; m < 8; ++m) inv[m] = 1.0f / lrow[m];
  __bf16* ob = Ot + (size_t)b * NSP * CH;
#pragma unroll
  for (int ct = 0; ct < 16; ++ct) {
#pragma unroll
    for (int m = 0; m < 8; ++m) {
      ob[(size_t)(q0 + m + half * 8) * CH + ct * 16 + l15] =
          (__bf16)(oacc[ct][m] * inv[m]);
    }
  }
}

// ---------------------------------------------------------------------------
// Output projection + bias + residual (fp32 out)
// ---------------------------------------------------------------------------
__global__ __launch_bounds__(128) void out_proj(const __bf16* __restrict__ Ot,
                                                const __bf16* __restrict__ Wo,
                                                const float*  __restrict__ bo,
                                                const float*  __restrict__ x,
                                                float* __restrict__ out) {
  const int lane = threadIdx.x & 31;
  const int l15  = lane & 15;
  const int half = lane >> 4;
  const int wave = blockIdx.x * (blockDim.x >> 5) + (threadIdx.x >> 5);
  const int nt   = wave & (NSP / 16 - 1);
  const int t    = wave >> 8;
  const int ot   = t & (CH / 16 - 1);
  const int b    = t >> 4;

  const __bf16* wr = Wo + (size_t)(ot * 16 + l15) * CH + half * 16;
  const __bf16* xb = Ot + (size_t)b * NSP * CH + (size_t)(nt * 16 + l15) * CH + half * 16;

  v8f acc = {};
#pragma unroll
  for (int c0 = 0; c0 < CH; c0 += 32) {
    v16bf a  = cat8(*(const v8bf*)(wr + c0), *(const v8bf*)(wr + c0 + 8));
    v16bf bb = *(const v16bf*)(xb + c0);
    acc = wmma_bf16(a, bb, acc);
  }

  const int obase = ot * 16 + half * 8;
  const int n     = nt * 16 + l15;
  const float* xr = x   + (size_t)b * CH * NSP + (size_t)obase * NSP + n;
  float*       op = out + (size_t)b * CH * NSP + (size_t)obase * NSP + n;
#pragma unroll
  for (int m = 0; m < 8; ++m)
    op[(size_t)m * NSP] = acc[m] + bo[obase + m] + xr[(size_t)m * NSP];
}

// ---------------------------------------------------------------------------
extern "C" void kernel_launch(void* const* d_in, const int* in_sizes, int n_in,
                              void* d_out, int out_size, void* d_ws, size_t ws_size,
                              hipStream_t stream) {
  const float* x   = (const float*)d_in[0];
  const float* ctx = (const float*)d_in[1];
  const float* wq  = (const float*)d_in[2];
  const float* bq  = (const float*)d_in[3];
  const float* wk  = (const float*)d_in[4];
  const float* bk  = (const float*)d_in[5];
  const float* wv  = (const float*)d_in[6];
  const float* bv  = (const float*)d_in[7];
  const float* wo  = (const float*)d_in[8];
  const float* bo  = (const float*)d_in[9];
  float* out = (float*)d_out;

  const size_t WSZ = (size_t)CH * CH;
  const size_t TSZ = (size_t)BATCH * NSP * CH;
  __bf16* wqb = (__bf16*)d_ws;
  __bf16* wkb = wqb + WSZ;
  __bf16* wvb = wkb + WSZ;
  __bf16* wob = wvb + WSZ;
  __bf16* Xt  = wob + WSZ;   // [B,N,C]; reused as Ot after Q projection
  __bf16* Ct  = Xt + TSZ;
  __bf16* Qt  = Ct + TSZ;
  __bf16* Kt  = Qt + TSZ;
  __bf16* Vm  = Kt + TSZ;
  __bf16* Ot  = Xt;

  cvt_f32_bf16<<<(int)(WSZ / 256), 256, 0, stream>>>(wq, wqb, (int)WSZ);
  cvt_f32_bf16<<<(int)(WSZ / 256), 256, 0, stream>>>(wk, wkb, (int)WSZ);
  cvt_f32_bf16<<<(int)(WSZ / 256), 256, 0, stream>>>(wv, wvb, (int)WSZ);
  cvt_f32_bf16<<<(int)(WSZ / 256), 256, 0, stream>>>(wo, wob, (int)WSZ);

  transpose_cvt<<<(int)(TSZ / 256), 256, 0, stream>>>(x, Xt);
  transpose_cvt<<<(int)(TSZ / 256), 256, 0, stream>>>(ctx, Ct);

  const int projBlocks = BATCH * (CH / 16) * (NSP / 16) / 4;  // 4096
  proj_gemm<1><<<projBlocks, 128, 0, stream>>>(Xt, wqb, bq, Qt);  // Q^T [B,N,C]
  proj_gemm<1><<<projBlocks, 128, 0, stream>>>(Ct, wkb, bk, Kt);  // K^T [B,N,C]
  proj_gemm<0><<<projBlocks, 128, 0, stream>>>(Ct, wvb, bv, Vm);  // V   [B,C,N]

  flash_attn<<<BATCH * (NSP / 16) / 4, 128, 0, stream>>>(Qt, Kt, Vm, Ot);

  out_proj<<<projBlocks, 128, 0, stream>>>(Ot, wob, bo, x, out);
}
